// SelfAttentionModule_65481071408314
// MI455X (gfx1250) — matrix operand order repaired and checked by reading, not compiled, not run on hardware
//
#include <hip/hip_runtime.h>
#include <hip/hip_bf16.h>

// ---------------------------------------------------------------------------
// Self-attention block for MI455X (gfx1250).
// bf16 WMMA for all matmuls; async global->LDS double-buffered GEMM staging;
// K-loops fully unrolled so accumulators stay pinned (no v_mov churn).
// ---------------------------------------------------------------------------

typedef __bf16 bf16_t;
typedef __attribute__((ext_vector_type(16))) __bf16 v16bf;
typedef __attribute__((ext_vector_type(8)))  float  v8f;

#define EMBED 1024
#define HEADS 16
#define HEAD_DIM 64
#define SEQ 2048
#define BATCH 2
#define MROWS (BATCH * SEQ)        // 4096 flattened token rows
#define NQKV (3 * EMBED)           // 3072

// ---- WMMA helpers ---------------------------------------------------------

__device__ __forceinline__ v8f wmma_bf16(v16bf a, v16bf b, v8f c) {
  // 8 args: (neg_a, A, neg_b, B, c_mod, C, reuse_a, reuse_b)
  return __builtin_amdgcn_wmma_f32_16x16x32_bf16(false, a, false, b,
                                                 (short)0, c, false, false);
}

// A-matrix fragment (16x32 bf16), loaded from a row-major [16 x >=32] tile.
// Lane L: row = L&15. Lanes 0-15 hold K={0..7,16..23}, lanes 16-31 hold
// K={8..15,24..31} (ISA 7.12.2 16-bit A table). Two b128 loads.
__device__ __forceinline__ v16bf load_fragA(const bf16_t* base, int ld, int koff) {
  int lane = threadIdx.x & 31;
  int row = lane & 15;
  int k0 = koff + ((lane & 16) ? 8 : 0);
  const bf16_t* p = base + row * ld + k0;
  union { v16bf v; uint4 u[2]; } t;
  t.u[0] = *reinterpret_cast<const uint4*>(p);
  t.u[1] = *reinterpret_cast<const uint4*>(p + 16);
  return t.v;
}

// B-matrix fragment (32x16 bf16) loaded from row-major B^T storage
// ([16 rows of n] x [K cols], stride ld). Lane L: column n = L&15.
// Lanes 0-15 hold K={0..7,16..23}, lanes 16-31 hold K={8..15,24..31}.
__device__ __forceinline__ v16bf load_fragB(const bf16_t* baseT, int ld, int koff) {
  int lane = threadIdx.x & 31;
  int n = lane & 15;
  int k0 = koff + ((lane & 16) ? 8 : 0);
  const bf16_t* p = baseT + n * ld + k0;
  union { v16bf v; uint4 u[2]; } t;
  t.u[0] = *reinterpret_cast<const uint4*>(p);
  t.u[1] = *reinterpret_cast<const uint4*>(p + 16);
  return t.v;
}

// ---- gfx1250 async global->LDS (ASYNCcnt path, bypasses VGPRs) ------------

__device__ __forceinline__ void async_ld_b128(bf16_t* lds_ptr, const bf16_t* gptr) {
  // Low 32 bits of a generic LDS pointer are the wave-relative LDS byte addr.
  unsigned int l = (unsigned int)(unsigned long long)lds_ptr;
  asm volatile("global_load_async_to_lds_b128 %0, %1, off"
               :: "v"(l), "v"(gptr)
               : "memory");
}

__device__ __forceinline__ void wait_async0() {
  asm volatile("s_wait_asynccnt 0" ::: "memory");
}

// ---- 1) fp32 -> bf16 conversion ------------------------------------------

__global__ void cvt_bf16_kernel(const float* __restrict__ in,
                                bf16_t* __restrict__ out, int n4) {
  int i = blockIdx.x * blockDim.x + threadIdx.x;
  if (i >= n4) return;
  float4 f = reinterpret_cast<const float4*>(in)[i];
  union { bf16_t h[4]; uint2 u; } o;
  o.h[0] = (bf16_t)f.x; o.h[1] = (bf16_t)f.y;
  o.h[2] = (bf16_t)f.z; o.h[3] = (bf16_t)f.w;
  reinterpret_cast<uint2*>(out)[i] = o.u;
}

// ---- GEMM tiling constants ------------------------------------------------

#define BM 128
#define BN 128
#define BK 32
#define KTILES (EMBED / BK)   // 32 K-steps, fully unrolled
#define LDT 40   // padded LDS row length (bf16); rows stay 16B aligned

// Stage one BMxBK (and BNxBK) tile into LDS with async b128 copies.
// 512 16-byte chunks per matrix, 256 threads -> 2 chunks each.
__device__ __forceinline__ void stage_tile_async(
    bf16_t* sA, bf16_t* sB,
    const bf16_t* __restrict__ A, const bf16_t* __restrict__ Bw,
    int gm0, int gn0, int kt) {
#pragma unroll
  for (int i = 0; i < 2; ++i) {
    int c = (int)threadIdx.x + i * 256;
    int row = c >> 2;
    int col = (c & 3) * 8;
    async_ld_b128(&sA[row * LDT + col], &A[(size_t)(gm0 + row) * EMBED + kt + col]);
    async_ld_b128(&sB[row * LDT + col], &Bw[(size_t)(gn0 + row) * EMBED + kt + col]);
  }
}

// Shared double-buffered WMMA mainloop: produces this wave's 4x2 accumulators.
__device__ __forceinline__ void gemm_mainloop(
    v8f acc[4][2], bf16_t (*sA)[BM * LDT], bf16_t (*sB)[BN * LDT],
    const bf16_t* __restrict__ A, const bf16_t* __restrict__ Bw,
    int gm0, int gn0, int wm, int wn) {
#pragma unroll
  for (int i = 0; i < 4; ++i)
#pragma unroll
    for (int j = 0; j < 2; ++j)
#pragma unroll
      for (int r = 0; r < 8; ++r) acc[i][j][r] = 0.f;

  stage_tile_async(sA[0], sB[0], A, Bw, gm0, gn0, 0);

#pragma unroll
  for (int t = 0; t < KTILES; ++t) {
    const int cur = t & 1;                // static after full unroll
    wait_async0();        // my async copies for buf[cur] are done
    __syncthreads();      // everyone's copies landed; prev compute finished
    if (t + 1 < KTILES)
      stage_tile_async(sA[cur ^ 1], sB[cur ^ 1], A, Bw, gm0, gn0, (t + 1) * BK);

    v16bf bf0 = load_fragB(&sB[cur][(wn + 0) * LDT], LDT, 0);
    v16bf bf1 = load_fragB(&sB[cur][(wn + 16) * LDT], LDT, 0);
#pragma unroll
    for (int sm = 0; sm < 4; ++sm) {
      v16bf af = load_fragA(&sA[cur][(wm + sm * 16) * LDT], LDT, 0);
      acc[sm][0] = wmma_bf16(af, bf0, acc[sm][0]);
      acc[sm][1] = wmma_bf16(af, bf1, acc[sm][1]);
    }
    // no trailing barrier: next iteration's wait+barrier precedes overwrite
  }
}

// ---- 2) QKV projection GEMM ----------------------------------------------
// qkv[m,n] = sum_k x[m,k] * Wqkv[n,k], scaled by attn_params[m,n].

__global__ __launch_bounds__(256) void qkv_gemm_kernel(
    const bf16_t* __restrict__ A, const bf16_t* __restrict__ Bw,
    const float* __restrict__ attn_params,
    bf16_t* __restrict__ Qb, bf16_t* __restrict__ Kb, bf16_t* __restrict__ VbT) {
  __shared__ bf16_t sA[2][BM * LDT];
  __shared__ bf16_t sB[2][BN * LDT];
  const int gm0 = blockIdx.y * BM;
  const int gn0 = blockIdx.x * BN;
  const int w = threadIdx.x >> 5;
  const int wm = (w & 1) * 64;
  const int wn = (w >> 1) * 32;

  v8f acc[4][2];
  gemm_mainloop(acc, sA, sB, A, Bw, gm0, gn0, wm, wn);

  // Epilogue: scale by attn_params, scatter into Q/K [B,H,S,D] and V^T [B,H,D,S].
  const int lane = threadIdx.x & 31;
  const int coln = lane & 15;
  const int rb = (lane & 16) ? 8 : 0;
#pragma unroll
  for (int sm = 0; sm < 4; ++sm)
#pragma unroll
    for (int sn = 0; sn < 2; ++sn)
#pragma unroll
      for (int r = 0; r < 8; ++r) {
        int m = gm0 + wm + sm * 16 + rb + r;
        int n = gn0 + wn + sn * 16 + coln;
        float v = acc[sm][sn][r] * attn_params[(size_t)m * NQKV + n];
        int b = m >> 11, s = m & (SEQ - 1);
        int seg = n >> 10, nn = n & (EMBED - 1);
        int h = nn >> 6, d = nn & 63;
        size_t headbase = ((size_t)(b * HEADS + h));
        if (seg == 0)
          Qb[(headbase * SEQ + s) * HEAD_DIM + d] = (bf16_t)v;
        else if (seg == 1)
          Kb[(headbase * SEQ + s) * HEAD_DIM + d] = (bf16_t)v;
        else
          VbT[(headbase * HEAD_DIM + d) * SEQ + s] = (bf16_t)v;
      }
}

// ---- 3) Flash attention ---------------------------------------------------
// One wave owns 16 q-rows of one (b,h); streams keys in chunks of 32.

__global__ __launch_bounds__(128) void attn_kernel(
    const bf16_t* __restrict__ Qb, const bf16_t* __restrict__ Kb,
    const bf16_t* __restrict__ VbT, bf16_t* __restrict__ Wt) {
  __shared__ bf16_t sP[4][16 * 32];   // per-wave P relayout scratch
  const int wv = threadIdx.x >> 5;
  const int gw = blockIdx.x * 4 + wv;
  const int per = SEQ / 16;           // 128 q-blocks per head
  const int bh = gw / per;
  const int qb = gw % per;

  const bf16_t* Qp = Qb + (size_t)bh * SEQ * HEAD_DIM + (size_t)qb * 16 * HEAD_DIM;
  const bf16_t* Kp = Kb + (size_t)bh * SEQ * HEAD_DIM;
  const bf16_t* Vt = VbT + (size_t)bh * HEAD_DIM * SEQ;

  v16bf aq0 = load_fragA(Qp, HEAD_DIM, 0);
  v16bf aq1 = load_fragA(Qp, HEAD_DIM, 32);

  float mrow[8], lrow[8];
  v8f acc[4];
#pragma unroll
  for (int r = 0; r < 8; ++r) { mrow[r] = -3.0e38f; lrow[r] = 0.f; }
#pragma unroll
  for (int t = 0; t < 4; ++t)
#pragma unroll
    for (int r = 0; r < 8; ++r) acc[t][r] = 0.f;

  const int lane = threadIdx.x & 31;
  const int coln = lane & 15;
  const int rb = (lane & 16) ? 8 : 0;
  bf16_t* Pl = sP[wv];

  for (int kk = 0; kk < SEQ; kk += 32) {
    // Prefetch next K/V chunk while we compute on this one (global_prefetch_b8).
    if (kk + 32 < SEQ) {
      const char* nk = (const char*)(Kp + (size_t)(kk + 32) * HEAD_DIM);
      __builtin_prefetch(nk + (size_t)lane * 128, 0, 1);
      __builtin_prefetch((const char*)(Vt + (size_t)lane * SEQ + kk + 32), 0, 1);
      __builtin_prefetch((const char*)(Vt + (size_t)(lane + 32) * SEQ + kk + 32), 0, 1);
    }

    v8f s0, s1;
#pragma unroll
    for (int r = 0; r < 8; ++r) { s0[r] = 0.f; s1[r] = 0.f; }
    v16bf bk00 = load_fragB(Kp + (size_t)kk * HEAD_DIM, HEAD_DIM, 0);
    v16bf bk01 = load_fragB(Kp + (size_t)kk * HEAD_DIM, HEAD_DIM, 32);
    v16bf bk10 = load_fragB(Kp + (size_t)(kk + 16) * HEAD_DIM, HEAD_DIM, 0);
    v16bf bk11 = load_fragB(Kp + (size_t)(kk + 16) * HEAD_DIM, HEAD_DIM, 32);
    s0 = wmma_bf16(aq0, bk00, s0);
    s0 = wmma_bf16(aq1, bk01, s0);
    s1 = wmma_bf16(aq0, bk10, s1);
    s1 = wmma_bf16(aq1, bk11, s1);

#pragma unroll
    for (int r = 0; r < 8; ++r) {
      float v0 = s0[r] * 0.125f;     // 1/sqrt(64)
      float v1 = s1[r] * 0.125f;
      float cm = fmaxf(v0, v1);
#pragma unroll
      for (int off = 1; off < 16; off <<= 1)
        cm = fmaxf(cm, __shfl_xor(cm, off, 16));
      float mn = fmaxf(mrow[r], cm);
      float sc = __expf(mrow[r] - mn);
      float p0 = __expf(v0 - mn);
      float p1 = __expf(v1 - mn);
      float ps = p0 + p1;
#pragma unroll
      for (int off = 1; off < 16; off <<= 1)
        ps += __shfl_xor(ps, off, 16);
      lrow[r] = lrow[r] * sc + ps;
      mrow[r] = mn;
#pragma unroll
      for (int t = 0; t < 4; ++t) acc[t][r] *= sc;
      int rr = rb + r;                      // C-layout row of this lane
      Pl[rr * 32 + coln]      = (bf16_t)p0;
      Pl[rr * 32 + 16 + coln] = (bf16_t)p1;
    }
    // P [16x32] back as A fragment (same-wave LDS ops are in order)
    v16bf ap = load_fragA(Pl, 32, 0);
#pragma unroll
    for (int t = 0; t < 4; ++t)
      acc[t] = wmma_bf16(ap, load_fragB(Vt + (size_t)(t * 16) * SEQ + kk, SEQ, 0), acc[t]);
  }

  // Normalize and store weighted [B,S,E] as bf16 for the output GEMM.
  const int b = bh / HEADS, h = bh % HEADS;
#pragma unroll
  for (int r = 0; r < 8; ++r) {
    float inv = 1.f / lrow[r];
    int srow = qb * 16 + rb + r;
    size_t base = ((size_t)(b * SEQ + srow)) * EMBED + h * HEAD_DIM;
#pragma unroll
    for (int t = 0; t < 4; ++t)
      Wt[base + t * 16 + coln] = (bf16_t)(acc[t][r] * inv);
  }
}

// ---- 4) Output projection GEMM + bias + residual -------------------------

__global__ __launch_bounds__(256) void out_gemm_kernel(
    const bf16_t* __restrict__ A, const bf16_t* __restrict__ Bw,
    const float* __restrict__ b_out, const float* __restrict__ x,
    float* __restrict__ pre) {
  __shared__ bf16_t sA[2][BM * LDT];
  __shared__ bf16_t sB[2][BN * LDT];
  const int gm0 = blockIdx.y * BM;
  const int gn0 = blockIdx.x * BN;
  const int w = threadIdx.x >> 5;
  const int wm = (w & 1) * 64;
  const int wn = (w >> 1) * 32;

  v8f acc[4][2];
  gemm_mainloop(acc, sA, sB, A, Bw, gm0, gn0, wm, wn);

  const int lane = threadIdx.x & 31;
  const int coln = lane & 15;
  const int rb = (lane & 16) ? 8 : 0;
#pragma unroll
  for (int sm = 0; sm < 4; ++sm)
#pragma unroll
    for (int sn = 0; sn < 2; ++sn)
#pragma unroll
      for (int r = 0; r < 8; ++r) {
        int m = gm0 + wm + sm * 16 + rb + r;
        int n = gn0 + wn + sn * 16 + coln;
        size_t idx = (size_t)m * EMBED + n;
        pre[idx] = acc[sm][sn][r] + b_out[n] + x[idx];
      }
}

// ---- 5) LayerNorm over last dim ------------------------------------------

__global__ __launch_bounds__(256) void ln_kernel(
    const float* __restrict__ pre, const float* __restrict__ gamma,
    const float* __restrict__ beta, float* __restrict__ out) {
  __shared__ float red[256];
  const int row = blockIdx.x;
  const float* p = pre + (size_t)row * EMBED;
  float sum = 0.f;
  for (int j = threadIdx.x; j < EMBED; j += 256) sum += p[j];
  red[threadIdx.x] = sum;
  __syncthreads();
  for (int s = 128; s > 0; s >>= 1) {
    if ((int)threadIdx.x < s) red[threadIdx.x] += red[threadIdx.x + s];
    __syncthreads();
  }
  float mu = red[0] * (1.0f / EMBED);
  __syncthreads();
  float sq = 0.f;
  for (int j = threadIdx.x; j < EMBED; j += 256) {
    float d = p[j] - mu;
    sq += d * d;
  }
  red[threadIdx.x] = sq;
  __syncthreads();
  for (int s = 128; s > 0; s >>= 1) {
    if ((int)threadIdx.x < s) red[threadIdx.x] += red[threadIdx.x + s];
    __syncthreads();
  }
  float rs = rsqrtf(red[0] * (1.0f / EMBED) + 1e-5f);
  for (int j = threadIdx.x; j < EMBED; j += 256)
    out[(size_t)row * EMBED + j] = (p[j] - mu) * rs * gamma[j] + beta[j];
}

// ---- host-side launcher ---------------------------------------------------

extern "C" void kernel_launch(void* const* d_in, const int* in_sizes, int n_in,
                              void* d_out, int out_size, void* d_ws, size_t ws_size,
                              hipStream_t stream) {
  const float* x     = (const float*)d_in[0];   // [B,S,E]
  const float* ap    = (const float*)d_in[1];   // [B,S,3E]
  const float* Wqkv  = (const float*)d_in[2];   // [3E,E]
  const float* Wout  = (const float*)d_in[3];   // [E,E]
  const float* b_out = (const float*)d_in[4];
  const float* gamma = (const float*)d_in[5];
  const float* beta  = (const float*)d_in[6];
  float* out = (float*)d_out;

  char* ws = (char*)d_ws;
  size_t off = 0;
  auto carve = [&](size_t bytes) {
    void* p = ws + off;
    off += (bytes + 255) & ~(size_t)255;
    return p;
  };
  bf16_t* xb    = (bf16_t*)carve((size_t)MROWS * EMBED * 2);   // 8 MiB
  bf16_t* Wqkvb = (bf16_t*)carve((size_t)NQKV * EMBED * 2);    // 6 MiB
  bf16_t* Woutb = (bf16_t*)carve((size_t)EMBED * EMBED * 2);   // 2 MiB
  bf16_t* Qb    = (bf16_t*)carve((size_t)MROWS * EMBED * 2);   // 8 MiB
  bf16_t* Kb    = (bf16_t*)carve((size_t)MROWS * EMBED * 2);   // 8 MiB
  bf16_t* VbT   = (bf16_t*)carve((size_t)MROWS * EMBED * 2);   // 8 MiB
  bf16_t* Wt    = (bf16_t*)carve((size_t)MROWS * EMBED * 2);   // 8 MiB
  float*  pre   = (float*)carve((size_t)MROWS * EMBED * 4);    // 16 MiB

  // 1) converts
  {
    int n4 = MROWS * EMBED / 4;
    cvt_bf16_kernel<<<(n4 + 255) / 256, 256, 0, stream>>>(x, xb, n4);
    n4 = NQKV * EMBED / 4;
    cvt_bf16_kernel<<<(n4 + 255) / 256, 256, 0, stream>>>(Wqkv, Wqkvb, n4);
    n4 = EMBED * EMBED / 4;
    cvt_bf16_kernel<<<(n4 + 255) / 256, 256, 0, stream>>>(Wout, Woutb, n4);
  }
  // 2) QKV projection
  qkv_gemm_kernel<<<dim3(NQKV / BN, MROWS / BM), 256, 0, stream>>>(
      xb, Wqkvb, ap, Qb, Kb, VbT);
  // 3) attention: BATCH*HEADS*(SEQ/16) wave-tasks, 4 waves per block
  attn_kernel<<<(BATCH * HEADS * (SEQ / 16)) / 4, 128, 0, stream>>>(Qb, Kb, VbT, Wt);
  // 4) output projection + residual
  out_gemm_kernel<<<dim3(EMBED / BN, MROWS / BM), 256, 0, stream>>>(
      Wt, Woutb, b_out, x, pre);
  // 5) layernorm
  ln_kernel<<<MROWS, 256, 0, stream>>>(pre, gamma, beta, out);
}
